// NoiScaleModule_21930103013990
// MI455X (gfx1250) — matrix operand order repaired and checked by reading, not compiled, not run on hardware
//
#include <hip/hip_runtime.h>

// ============================================================================
// MI455X / gfx1250 implementation.
//
// Roofline: ~70 GFLOP total, ~300-400 MB HBM traffic (~15us @ 23.3 TB/s).
// All large GEMMs use v_wmma_f32_16x16x32_f16 (f16 A/B, f32 accumulate);
// weights are converted to f16 once per launch, activations are converted
// f32->f16 while staging into LDS (one rounding per operand).
// rfft / irfft are expressed as DFT GEMMs with f16 twiddle matrices so the
// frequency branch also rides the WMMA path.
// Dead code eliminated: xcat = concat(low, high[...,256:]) == low, so the
// whole "high" branch of ttf_freqaug is skipped.
// Small latency-bound ops (32x32 attention, complex attention, layernorms)
// run in f32 VALU.
//
// GEMM staging:
//  - B tile (f16 weights, no conversion needed): gfx1250 async global->LDS
//    copies (global_load_async_to_lds_b128, ASYNCcnt) -- no VGPR round trip,
//    overlaps with the A-tile convert pipeline.
//  - A tile (f32 activations): float4 loads + v_cvt_pk_f16_f32 + ds_store.
//  - branchless clamped slow path for the N=257 / K-tail edge tiles.
// xf buffers are padded ld=272 so all fast-path vector loads stay aligned.
//
// Workspace requirement: ~91 MB (15.3 MB f16 weights + 75.6 MB f32 buffers).
// ============================================================================

typedef _Float16 half_t;
typedef __attribute__((ext_vector_type(16))) _Float16 v16h;
typedef __attribute__((ext_vector_type(8)))  _Float16 v8h;
typedef __attribute__((ext_vector_type(4)))  _Float16 v4h;
typedef __attribute__((ext_vector_type(8)))  float    v8f;

#define BM 128
#define BN 128
#define BK 32
#define LDS_STRIDE 40  // 32 + 8 halves pad -> conflict-free 16B LDS reads

// ---------------------------------------------------------------------------
// WMMA GEMM: C[M,N](f32) = act( alpha * A[M,K](f32->f16) @ B[N,K]^T(f16)
//                               + (RES ? Res : 0) + (BIAS ? bias[n] : 0) )
// B is stored row-major [N,K] (torch weight layout w[N,K], y = x @ w.T).
// ---------------------------------------------------------------------------
template<bool RELU, bool BIAS, bool RES>
__global__ __launch_bounds__(256) void k_gemm(
    const float* __restrict__ Aptr, int lda,
    const half_t* __restrict__ Bptr, int ldb,
    float* Cptr, int ldc,
    const float* __restrict__ bias,
    const float* Res,
    int M, int N, int K, float alpha)
{
  __shared__ __align__(16) half_t As[BM * LDS_STRIDE];
  __shared__ __align__(16) half_t Bs[BN * LDS_STRIDE];

  const int tid  = threadIdx.x;
  const int lane = tid & 31;
  const int wave = tid >> 5;     // 8 waves
  const int wm   = wave >> 2;    // 0..1 : 64-row band
  const int wn   = wave & 3;     // 0..3 : 32-col band
  const int g    = lane >> 4;    // lane half-group
  const int l15  = lane & 15;
  const int bm0  = blockIdx.y * BM;
  const int bn0  = blockIdx.x * BN;

  v8f acc[4][2] = {};

  // Fast path legal when the whole tile is in-bounds and rows are vector-
  // aligned (guaranteed by construction for all non-edge blocks here).
  const bool mn_fast = (bm0 + BM <= M) && (bn0 + BN <= N) &&
                       ((lda & 3) == 0) && ((ldb & 7) == 0);

  for (int k0 = 0; k0 < K; k0 += BK) {
    if (mn_fast && (k0 + BK <= K)) {
      // ---- B tile: async global->LDS (16B per lane, ASYNCcnt tracked).
      // LDS dest offset = low 32 bits of the flat shared-mem address
      // (ISA 10.2: LDS aperture addresses map via addr[31:0]).
      #pragma unroll
      for (int i = 0; i < 2; ++i) {
        int idx = tid + i * 256;            // 0..511 v8h slots (128x4)
        int r = idx >> 2, c8 = (idx & 3) << 3;
        unsigned ldsoff = (unsigned)(size_t)&Bs[r * LDS_STRIDE + c8];
        const half_t* ga = &Bptr[(size_t)(bn0 + r) * ldb + k0 + c8];
        asm volatile("global_load_async_to_lds_b128 %0, %1, off"
                     :: "v"(ldsoff), "v"(ga) : "memory");
      }
      // ---- A tile: issue all f32 vector loads, then convert + LDS store.
      float4 a4[4];
      #pragma unroll
      for (int i = 0; i < 4; ++i) {
        int idx = tid + i * 256;            // 0..1023 float4 slots (128x8)
        int r = idx >> 3, c4 = (idx & 7) << 2;
        a4[i] = *(const float4*)&Aptr[(size_t)(bm0 + r) * lda + k0 + c4];
      }
      #pragma unroll
      for (int i = 0; i < 4; ++i) {
        int idx = tid + i * 256;
        int r = idx >> 3, c4 = (idx & 7) << 2;
        v4h h;
        h[0] = (half_t)a4[i].x; h[1] = (half_t)a4[i].y;
        h[2] = (half_t)a4[i].z; h[3] = (half_t)a4[i].w;
        *(v4h*)&As[r * LDS_STRIDE + c4] = h;
      }
      // All of this wave's async B copies must land before the barrier.
      asm volatile("s_wait_asynccnt 0x0" ::: "memory");
    } else {
      // ---- branchless clamped slow path (edge tiles only)
      #pragma unroll
      for (int i = 0; i < 16; ++i) {
        int idx = tid + i * 256;            // 0..4095
        int r = idx >> 5, c = idx & 31;
        int gr = bm0 + r, gc = k0 + c, gn = bn0 + r;
        bool aok = (gr < M) && (gc < K);
        bool bok = (gn < N) && (gc < K);
        int grc = gr < M ? gr : M - 1;
        int gnc = gn < N ? gn : N - 1;
        int gcc = gc < K ? gc : K - 1;
        float  av = Aptr[(size_t)grc * lda + gcc];
        half_t bv = Bptr[(size_t)gnc * ldb + gcc];
        As[r * LDS_STRIDE + c] = aok ? (half_t)av : (half_t)0.0f;
        Bs[r * LDS_STRIDE + c] = bok ? bv : (half_t)0.0f;
      }
    }
    __syncthreads();

    // Fragments per ISA 7.12.2 layouts.
    v16h aF[4], bF[2];
    #pragma unroll
    for (int t = 0; t < 4; ++t) {
      int row = wm * 64 + t * 16 + l15;
      // A 16x32 f16: lane-half g holds K = {8g..8g+7, 16+8g..23+8g}
      v8h lo = *(const v8h*)&As[row * LDS_STRIDE + 8 * g];
      v8h hi = *(const v8h*)&As[row * LDS_STRIDE + 16 + 8 * g];
      aF[t] = __builtin_shufflevector(lo, hi, 0,1,2,3,4,5,6,7,8,9,10,11,12,13,14,15);
    }
    #pragma unroll
    for (int u = 0; u < 2; ++u) {
      int rn = wn * 32 + u * 16 + l15;
      // B 32x16 f16: lane-half g holds K = {8g..8g+7, 16+8g..23+8g}
      v8h lo = *(const v8h*)&Bs[rn * LDS_STRIDE + 8 * g];
      v8h hi = *(const v8h*)&Bs[rn * LDS_STRIDE + 16 + 8 * g];
      bF[u] = __builtin_shufflevector(lo, hi, 0,1,2,3,4,5,6,7,8,9,10,11,12,13,14,15);
    }
    #pragma unroll
    for (int t = 0; t < 4; ++t)
      #pragma unroll
      for (int u = 0; u < 2; ++u)
        acc[t][u] = __builtin_amdgcn_wmma_f32_16x16x32_f16(
            false, aF[t], false, bF[u], (short)0, acc[t][u], false, false);
    __syncthreads();
  }

  // Epilogue: C/D layout -> m_local = g*8 + r, n_local = l15
  #pragma unroll
  for (int t = 0; t < 4; ++t) {
    #pragma unroll
    for (int u = 0; u < 2; ++u) {
      int n = bn0 + wn * 32 + u * 16 + l15;
      if (n >= N) continue;
      int mbase = bm0 + wm * 64 + t * 16 + g * 8;
      float badd = BIAS ? bias[n] : 0.0f;
      #pragma unroll
      for (int r8 = 0; r8 < 8; ++r8) {
        int m = mbase + r8;
        if (m < M) {
          float v = alpha * acc[t][u][r8];
          if (RES) v += Res[(size_t)m * ldc + n];
          v += badd;
          if (RELU) v = fmaxf(v, 0.0f);
          Cptr[(size_t)m * ldc + n] = v;
        }
      }
    }
  }
}

// ---------------------------------------------------------------------------
// Multi-head attention core: one block per (batch b, head h). seq=32, dh=64.
// QKV layout: [B*32, 1536] (q | k | v each 512 wide). O: [B*32, 512].
// ---------------------------------------------------------------------------
__global__ __launch_bounds__(256) void k_attn(const float* __restrict__ QKV,
                                              float* __restrict__ O)
{
  int b = blockIdx.x >> 3, h = blockIdx.x & 7;
  __shared__ float qs[32][64], ks[32][64], vs[32][64], sc[32][33];
  int tid = threadIdx.x;
  const float* base = QKV + (size_t)b * 32 * 1536;
  for (int i = tid; i < 32 * 64; i += 256) {
    int s = i >> 6, d = i & 63;
    qs[s][d] = base[s * 1536 + h * 64 + d];
    ks[s][d] = base[s * 1536 + 512 + h * 64 + d];
    vs[s][d] = base[s * 1536 + 1024 + h * 64 + d];
  }
  __syncthreads();
  for (int i = tid; i < 1024; i += 256) {
    int q = i >> 5, k = i & 31;
    float sum = 0.0f;
    #pragma unroll 8
    for (int d = 0; d < 64; ++d) sum += qs[q][d] * ks[k][d];
    sc[q][k] = sum * 0.125f;  // 1/sqrt(64)
  }
  __syncthreads();
  if (tid < 32) {
    float mx = -1e30f;
    for (int k = 0; k < 32; ++k) mx = fmaxf(mx, sc[tid][k]);
    float sum = 0.0f;
    for (int k = 0; k < 32; ++k) { float e = __expf(sc[tid][k] - mx); sc[tid][k] = e; sum += e; }
    float inv = 1.0f / sum;
    for (int k = 0; k < 32; ++k) sc[tid][k] *= inv;
  }
  __syncthreads();
  for (int i = tid; i < 32 * 64; i += 256) {
    int q = i >> 6, d = i & 63;
    float sum = 0.0f;
    #pragma unroll 8
    for (int k = 0; k < 32; ++k) sum += sc[q][k] * vs[k][d];
    O[((size_t)b * 32 + q) * 512 + h * 64 + d] = sum;
  }
}

// ---------------------------------------------------------------------------
// Row layernorm with fused residual: Y[row] = LN(X[row] + R[row]) * w + b.
// One block per row of 512.
// ---------------------------------------------------------------------------
__global__ __launch_bounds__(256) void k_ln_row(const float* __restrict__ X,
                                                const float* __restrict__ R,
                                                const float* __restrict__ w,
                                                const float* __restrict__ bb,
                                                float* __restrict__ Y)
{
  int row = blockIdx.x, tid = threadIdx.x;
  const float* x = X + (size_t)row * 512;
  const float* r = R + (size_t)row * 512;
  float v0 = x[tid] + r[tid];
  float v1 = x[tid + 256] + r[tid + 256];
  __shared__ float red[256];
  red[tid] = v0 + v1;
  __syncthreads();
  for (int st = 128; st > 0; st >>= 1) { if (tid < st) red[tid] += red[tid + st]; __syncthreads(); }
  float mean = red[0] * (1.0f / 512.0f);
  __syncthreads();
  float d0 = v0 - mean, d1 = v1 - mean;
  red[tid] = d0 * d0 + d1 * d1;
  __syncthreads();
  for (int st = 128; st > 0; st >>= 1) { if (tid < st) red[tid] += red[tid + st]; __syncthreads(); }
  float rstd = rsqrtf(red[0] * (1.0f / 512.0f) + 1e-5f);
  Y[(size_t)row * 512 + tid]       = d0 * rstd * w[tid]       + bb[tid];
  Y[(size_t)row * 512 + tid + 256] = d1 * rstd * w[tid + 256] + bb[tid + 256];
}

// ---------------------------------------------------------------------------
// Global layernorm over (C,S)=(32,512) per batch with fused input ReLU:
// Y[b] = LN_{C,S}(relu(X[b])) * w + b.  One block per batch sample.
// ---------------------------------------------------------------------------
__global__ __launch_bounds__(256) void k_gln_relu(const float* __restrict__ X,
                                                  const float* __restrict__ w,
                                                  const float* __restrict__ bp,
                                                  float* __restrict__ Y)
{
  int bidx = blockIdx.x, tid = threadIdx.x;
  const float* x = X + (size_t)bidx * 16384;
  float s = 0.0f, sq = 0.0f;
  for (int i = tid; i < 16384; i += 256) {
    float v = fmaxf(x[i], 0.0f);
    s += v; sq += v * v;
  }
  __shared__ float r1[256], r2[256];
  r1[tid] = s; r2[tid] = sq;
  __syncthreads();
  for (int st = 128; st > 0; st >>= 1) {
    if (tid < st) { r1[tid] += r1[tid + st]; r2[tid] += r2[tid + st]; }
    __syncthreads();
  }
  float mean = r1[0] * (1.0f / 16384.0f);
  float var  = r2[0] * (1.0f / 16384.0f) - mean * mean;
  float rstd = rsqrtf(var + 1e-5f);
  for (int i = tid; i < 16384; i += 256) {
    float v = fmaxf(x[i], 0.0f);
    Y[(size_t)bidx * 16384 + i] = (v - mean) * rstd * w[i] + bp[i];
  }
}

// ---------------------------------------------------------------------------
// Complex self-attention (Q=K=V=low), out = low + softmax_c(Q Q^T / 16) Q.
// Softmax applied to re and im parts independently. One block per batch b;
// low: [B*32, 256] complex as separate re/im planes.
// ---------------------------------------------------------------------------
__global__ __launch_bounds__(256) void k_cattn(const float* __restrict__ lr,
                                               const float* __restrict__ li,
                                               float* __restrict__ outr,
                                               float* __restrict__ outi)
{
  int b = blockIdx.x, tid = threadIdx.x;
  const float* qr = lr + (size_t)b * 32 * 256;
  const float* qi = li + (size_t)b * 32 * 256;
  __shared__ float scr[32][33], sci[32][33];
  for (int i = tid; i < 1024; i += 256) {
    int q = i >> 5, k = i & 31;
    float sr = 0.0f, si = 0.0f;
    for (int e = 0; e < 256; ++e) {
      float ar = qr[q * 256 + e], ai = qi[q * 256 + e];
      float br = qr[k * 256 + e], bi = qi[k * 256 + e];
      sr += ar * br - ai * bi;
      si += ar * bi + ai * br;
    }
    scr[q][k] = sr * 0.0625f;  // 1/sqrt(256)
    sci[q][k] = si * 0.0625f;
  }
  __syncthreads();
  if (tid < 64) {
    int q = tid >> 1;
    float* row = (tid & 1) ? sci[q] : scr[q];
    float mx = -1e30f;
    for (int k = 0; k < 32; ++k) mx = fmaxf(mx, row[k]);
    float sum = 0.0f;
    for (int k = 0; k < 32; ++k) { float e = __expf(row[k] - mx); row[k] = e; sum += e; }
    float inv = 1.0f / sum;
    for (int k = 0; k < 32; ++k) row[k] *= inv;
  }
  __syncthreads();
  for (int i = tid; i < 32 * 256; i += 256) {
    int q = i >> 8, e = i & 255;
    float sr = 0.0f, si = 0.0f;
    #pragma unroll 8
    for (int k = 0; k < 32; ++k) {
      float wr = scr[q][k], wi = sci[q][k];
      float vr = qr[k * 256 + e], vi = qi[k * 256 + e];
      sr += wr * vr - wi * vi;
      si += wr * vi + wi * vr;
    }
    outr[(size_t)b * 8192 + i] = qr[q * 256 + e] + sr;
    outi[(size_t)b * 8192 + i] = qi[q * 256 + e] + si;
  }
}

// ---------------------------------------------------------------------------
// Elementwise / prep kernels
// ---------------------------------------------------------------------------
__global__ void k_cvt(const float* __restrict__ in, half_t* __restrict__ out, int n) {
  int i = blockIdx.x * 256 + threadIdx.x;
  if (i < n) out[i] = (half_t)in[i];
}
// l1 [128,256] (K,N) -> [256,128] (N,K) f16
__global__ void k_tcvt_l1(const float* __restrict__ in, half_t* __restrict__ out) {
  int i = blockIdx.x * 256 + threadIdx.x;
  if (i < 32768) { int n = i >> 7, k = i & 127; out[i] = (half_t)in[k * 256 + n]; }
}
__global__ void k_copy(const float* __restrict__ in, float* __restrict__ out, int n) {
  int i = blockIdx.x * 256 + threadIdx.x;
  if (i < n) out[i] = in[i];
}
__global__ void k_add(const float* __restrict__ a, const float* __restrict__ b,
                      float* __restrict__ out, int n) {
  int i = blockIdx.x * 256 + threadIdx.x;
  if (i < n) out[i] = a[i] + b[i];
}
// Forward rDFT (ortho): Fr[k,s] = cos(2pi k s/512)/sqrt(512), Fi = -sin(..)/sqrt(512)
__global__ void k_build_rdft(half_t* __restrict__ Fr, half_t* __restrict__ Fi) {
  int i = blockIdx.x * 256 + threadIdx.x;
  if (i >= 257 * 512) return;
  int k = i >> 9, s = i & 511;
  int ph = (k * s) & 511;
  float ang = -6.283185307179586f * (float)ph * (1.0f / 512.0f);
  float sn, cs; __sincosf(ang, &sn, &cs);
  const float inv = 0.04419417382415922f;  // 1/sqrt(512)
  Fr[i] = (half_t)(cs * inv);
  Fi[i] = (half_t)(sn * inv);
}
// Inverse rDFT (ortho, Nyquist bin absent/zero):
// Wr[s,k] = wk*cos(2pi k s/512), Wi[s,k] = -wk*sin(..), wk=(k==0?1:2)/sqrt(512)
__global__ void k_build_irdft(half_t* __restrict__ Wr, half_t* __restrict__ Wi) {
  int i = blockIdx.x * 256 + threadIdx.x;
  if (i >= 512 * 256) return;
  int s = i >> 8, k = i & 255;
  int ph = (k * s) & 511;
  float ang = 6.283185307179586f * (float)ph * (1.0f / 512.0f);
  float sn, cs; __sincosf(ang, &sn, &cs);
  float scale = (k == 0 ? 1.0f : 2.0f) * 0.04419417382415922f;
  Wr[i] = (half_t)(cs * scale);
  Wi[i] = (half_t)(-sn * scale);
}

// ---------------------------------------------------------------------------
// Host-side helpers
// ---------------------------------------------------------------------------
static inline void gemm(hipStream_t st,
                        const float* Aptr, int lda,
                        const half_t* Bptr, int ldb,
                        float* Cptr, int ldc,
                        const float* bias, const float* res,
                        int M, int N, int K, float alpha, bool relu)
{
  dim3 g((unsigned)((N + BN - 1) / BN), (unsigned)((M + BM - 1) / BM));
  dim3 blk(256);
#define LG(R_, B_, S_) k_gemm<R_, B_, S_><<<g, blk, 0, st>>>(Aptr, lda, Bptr, ldb, Cptr, ldc, bias, res, M, N, K, alpha)
  int sel = (relu ? 4 : 0) | (bias ? 2 : 0) | (res ? 1 : 0);
  switch (sel) {
    case 0: LG(false, false, false); break;
    case 1: LG(false, false, true);  break;
    case 2: LG(false, true,  false); break;
    case 3: LG(false, true,  true);  break;
    case 4: LG(true,  false, false); break;
    case 5: LG(true,  false, true);  break;
    case 6: LG(true,  true,  false); break;
    case 7: LG(true,  true,  true);  break;
  }
#undef LG
}

// One TransformerEncoderLayer (post-norm, relu, batch_first).
static void run_tel(hipStream_t st, const float* Xin, float* Xout,
                    const half_t* w_in, const float* b_in,
                    const half_t* w_out, const float* b_out,
                    const float* ln1w, const float* ln1b,
                    const half_t* w_l1, const float* b_l1,
                    const half_t* w_l2, const float* b_l2,
                    const float* ln2w, const float* ln2b,
                    float* bufBig, float* bufO, float* bufT, float* bufX1)
{
  gemm(st, Xin, 512, w_in, 512, bufBig, 1536, b_in, nullptr, 4096, 1536, 512, 1.0f, false);
  k_attn<<<1024, 256, 0, st>>>(bufBig, bufO);
  gemm(st, bufO, 512, w_out, 512, bufT, 512, b_out, nullptr, 4096, 512, 512, 1.0f, false);
  k_ln_row<<<4096, 256, 0, st>>>(Xin, bufT, ln1w, ln1b, bufX1);
  gemm(st, bufX1, 512, w_l1, 512, bufBig, 2048, b_l1, nullptr, 4096, 2048, 512, 1.0f, true);
  gemm(st, bufBig, 2048, w_l2, 2048, bufT, 512, b_l2, nullptr, 4096, 512, 2048, 1.0f, false);
  k_ln_row<<<4096, 256, 0, st>>>(bufX1, bufT, ln2w, ln2b, Xout);
}

extern "C" void kernel_launch(void* const* d_in, const int* in_sizes, int n_in,
                              void* d_out, int out_size, void* d_ws, size_t ws_size,
                              hipStream_t stream)
{
  (void)in_sizes; (void)n_in; (void)out_size; (void)ws_size;
  // ----- inputs (setup_inputs order) -----
  const float* x = (const float*)d_in[0];
  const float* t_in_w  = (const float*)d_in[1];  const float* t_in_b  = (const float*)d_in[2];
  const float* t_out_w = (const float*)d_in[3];  const float* t_out_b = (const float*)d_in[4];
  const float* t_ln1_w = (const float*)d_in[5];  const float* t_ln1_b = (const float*)d_in[6];
  const float* t_l1_w  = (const float*)d_in[7];  const float* t_l1_b  = (const float*)d_in[8];
  const float* t_l2_w  = (const float*)d_in[9];  const float* t_l2_b  = (const float*)d_in[10];
  const float* t_ln2_w = (const float*)d_in[11]; const float* t_ln2_b = (const float*)d_in[12];
  const float* r_in_w  = (const float*)d_in[13]; const float* r_in_b  = (const float*)d_in[14];
  const float* r_out_w = (const float*)d_in[15]; const float* r_out_b = (const float*)d_in[16];
  const float* r_ln1_w = (const float*)d_in[17]; const float* r_ln1_b = (const float*)d_in[18];
  const float* r_l1_w  = (const float*)d_in[19]; const float* r_l1_b  = (const float*)d_in[20];
  const float* r_l2_w  = (const float*)d_in[21]; const float* r_l2_b  = (const float*)d_in[22];
  const float* r_ln2_w = (const float*)d_in[23]; const float* r_ln2_b = (const float*)d_in[24];
  const float* lc_w  = (const float*)d_in[25]; const float* lc_b  = (const float*)d_in[26];
  const float* up1_w = (const float*)d_in[27]; const float* up1_b = (const float*)d_in[28];
  const float* up2_w = (const float*)d_in[29]; const float* up2_b = (const float*)d_in[30];
  const float* n1_w = (const float*)d_in[31]; const float* n1_b = (const float*)d_in[32];
  const float* n2_w = (const float*)d_in[33]; const float* n2_b = (const float*)d_in[34];
  const float* l1_re = (const float*)d_in[35]; const float* l1_im = (const float*)d_in[36];
  // d_in[37], d_in[38]: h1_re / h1_im  -- dead (high branch unused)
  const float* lb1_re = (const float*)d_in[39]; const float* lb1_im = (const float*)d_in[40];
  // d_in[41], d_in[42]: hb1_re / hb1_im -- dead

  // ----- outputs: [res | down_x | fa | up], each 4096*512 f32 -----
  float* out_res = (float*)d_out;
  float* out_dx  = out_res + 2097152;
  float* out_fa  = out_res + 4194304;
  float* out_up  = out_res + 6291456;

  // ----- workspace layout -----
  half_t* hw = (half_t*)d_ws;
  half_t* h_t_in  = hw + 0;        // 1536*512
  half_t* h_t_out = hw + 786432;   // 512*512
  half_t* h_t_l1  = hw + 1048576;  // 2048*512
  half_t* h_t_l2  = hw + 2097152;  // 512*2048
  half_t* h_r_in  = hw + 3145728;
  half_t* h_r_out = hw + 3932160;
  half_t* h_r_l1  = hw + 4194304;
  half_t* h_r_l2  = hw + 5242880;
  half_t* h_lc    = hw + 6291456;  // 512*512
  half_t* h_up1   = hw + 6553600;
  half_t* h_up2   = hw + 6815744;
  half_t* h_l1r   = hw + 7077888;  // 256*128
  half_t* h_l1i   = hw + 7110656;
  half_t* h_Fr    = hw + 7143424;  // 257*512
  half_t* h_Fi    = hw + 7275008;
  half_t* h_Wr    = hw + 7406592;  // 512*256
  half_t* h_Wi    = hw + 7537664;  // ends at 7668736 halves = 15337472 bytes

  float* fw = (float*)((char*)d_ws + 15337472);
  float* bufBig = fw + 0;          // 4096*2048 (QKV / FFN hidden / freq overlay)
  float* bufO   = fw + 8388608;    // 4096*512
  float* bufT   = fw + 10485760;   // 4096*512
  float* bufX1  = fw + 12582912;   // 4096*512
  float* bufTel = fw + 14680064;   // 4096*512
  float* bufDf  = fw + 16777216;   // 4096*512  (lives until final add)
  // freq-branch overlay inside bufBig (only live between the two TELs).
  // xf padded to ld=272 so fast-path float4 loads stay 16B-aligned.
  float* xf_r  = bufBig + 0;        // 4096*272
  float* xf_i  = bufBig + 1114112;  // 4096*272
  float* cl_r  = bufBig + 2228224;  // 4096*256
  float* cl_i  = bufBig + 3276800;
  float* cl2_r = bufBig + 4325376;
  float* cl2_i = bufBig + 5373952;  // ends 6422528 < 8388608

  dim3 blk(256);
  auto grid1 = [](int n) { return dim3((unsigned)((n + 255) / 256)); };

  // ----- weight conversion + twiddle build -----
  k_cvt<<<grid1(786432), blk, 0, stream>>>(t_in_w, h_t_in, 786432);
  k_cvt<<<grid1(262144), blk, 0, stream>>>(t_out_w, h_t_out, 262144);
  k_cvt<<<grid1(1048576), blk, 0, stream>>>(t_l1_w, h_t_l1, 1048576);
  k_cvt<<<grid1(1048576), blk, 0, stream>>>(t_l2_w, h_t_l2, 1048576);
  k_cvt<<<grid1(786432), blk, 0, stream>>>(r_in_w, h_r_in, 786432);
  k_cvt<<<grid1(262144), blk, 0, stream>>>(r_out_w, h_r_out, 262144);
  k_cvt<<<grid1(1048576), blk, 0, stream>>>(r_l1_w, h_r_l1, 1048576);
  k_cvt<<<grid1(1048576), blk, 0, stream>>>(r_l2_w, h_r_l2, 1048576);
  k_cvt<<<grid1(262144), blk, 0, stream>>>(lc_w, h_lc, 262144);
  k_cvt<<<grid1(262144), blk, 0, stream>>>(up1_w, h_up1, 262144);
  k_cvt<<<grid1(262144), blk, 0, stream>>>(up2_w, h_up2, 262144);
  k_tcvt_l1<<<grid1(32768), blk, 0, stream>>>(l1_re, h_l1r);
  k_tcvt_l1<<<grid1(32768), blk, 0, stream>>>(l1_im, h_l1i);
  k_build_rdft<<<grid1(257 * 512), blk, 0, stream>>>(h_Fr, h_Fi);
  k_build_irdft<<<grid1(512 * 256), blk, 0, stream>>>(h_Wr, h_Wi);

  // down_x output = identity copy of x
  k_copy<<<grid1(2097152), blk, 0, stream>>>(x, out_dx, 2097152);

  // ----- main branch: TEL(x) -> relu -> global LN -> df -----
  run_tel(stream, x, bufTel,
          h_t_in, t_in_b, h_t_out, t_out_b, t_ln1_w, t_ln1_b,
          h_t_l1, t_l1_b, h_t_l2, t_l2_b, t_ln2_w, t_ln2_b,
          bufBig, bufO, bufT, bufX1);
  k_gln_relu<<<128, 256, 0, stream>>>(bufTel, n1_w, n1_b, bufDf);

  // ----- frequency branch (rfft -> complex linear+crelu -> cattn -> irfft) --
  // rfft as DFT GEMMs: xf = x @ F^T   (N = 257 bins, ld padded to 272)
  gemm(stream, x, 512, h_Fr, 512, xf_r, 272, nullptr, nullptr, 4096, 257, 512, 1.0f, false);
  gemm(stream, x, 512, h_Fi, 512, xf_i, 272, nullptr, nullptr, 4096, 257, 512, 1.0f, false);
  // low = crelu(xf[:, :128] @ l1 + lb1)  (complex GEMM via 4 real GEMMs)
  gemm(stream, xf_r, 272, h_l1r, 128, cl_r, 256, nullptr, nullptr, 4096, 256, 128, 1.0f, false);
  gemm(stream, xf_i, 272, h_l1i, 128, cl_r, 256, lb1_re, cl_r, 4096, 256, 128, -1.0f, true);
  gemm(stream, xf_r, 272, h_l1i, 128, cl_i, 256, nullptr, nullptr, 4096, 256, 128, 1.0f, false);
  gemm(stream, xf_i, 272, h_l1r, 128, cl_i, 256, lb1_im, cl_i, 4096, 256, 128, 1.0f, true);
  // low = low + cattn(low)
  k_cattn<<<128, 256, 0, stream>>>(cl_r, cl_i, cl2_r, cl2_i);
  // irfft (xcat == low; Nyquist bin zero): fa_raw = cl2_r @ Wr^T + cl2_i @ Wi^T
  gemm(stream, cl2_r, 256, h_Wr, 256, bufT, 512, nullptr, nullptr, 4096, 512, 256, 1.0f, false);
  gemm(stream, cl2_i, 256, h_Wi, 256, bufT, 512, nullptr, bufT, 4096, 512, 256, 1.0f, false);
  // fa = fa_raw @ lc_w^T + lc_b   (written straight into the fa output slot)
  gemm(stream, bufT, 512, h_lc, 512, out_fa, 512, lc_b, nullptr, 4096, 512, 512, 1.0f, false);
  // up = (fa @ up1^T + b) @ up2^T + b
  gemm(stream, out_fa, 512, h_up1, 512, bufO, 512, up1_b, nullptr, 4096, 512, 512, 1.0f, false);
  gemm(stream, bufO, 512, h_up2, 512, out_up, 512, up2_b, nullptr, 4096, 512, 512, 1.0f, false);

  // ----- recons branch: TEL(fa) -> relu -> global LN -> rf -----
  run_tel(stream, out_fa, bufTel,
          h_r_in, r_in_b, h_r_out, r_out_b, r_ln1_w, r_ln1_b,
          h_r_l1, r_l1_b, h_r_l2, r_l2_b, r_ln2_w, r_ln2_b,
          bufBig, bufO, bufT, bufX1);
  k_gln_relu<<<128, 256, 0, stream>>>(bufTel, n2_w, n2_b, bufX1);

  // res = df + rf
  k_add<<<grid1(2097152), blk, 0, stream>>>(bufDf, bufX1, out_res, 2097152);
}
